// CON_GATLayer_68384469286918
// MI455X (gfx1250) — matrix operand order repaired and checked by reading, not compiled, not run on hardware
//
#include <hip/hip_runtime.h>
#include <hip/hip_bf16.h>
#include <stdint.h>

// ---------------------------------------------------------------------------
// CON_GATLayer fused kernel for MI455X (gfx1250, wave32, WMMA)
//
// N=3072, IN=256, ATT_H=64, DV=32, H=4.
// Stage 1: 6 projection GEMMs (x @ W + b) -> bf16 Q/K (H,N,64), V^T (H,32,N)
// Stage 2: fused attention per (head, 16-row strip):
//   phase 1 : att1/att2 strips (16x3072) via v_wmma_f32_16x16x32_bf16 -> LDS
//   phase 2a: LDS-only row-max bound M[i] = rowmax(att1)+rowmax(att2)
//             (valid shift for BOTH branches: gather stays within the row,
//              mask <= 0) -- avoids a whole global pass over idx/graphs
//   phase 2b: P = exp(l - M) rebuilt per fragment (bf16), P @ V via WMMA,
//             row sums accumulated per-lane in f32 (fragment elements cover
//             disjoint j), reduced with ds_add_f32; final divide normalizes.
//             Next-chunk graph/index lines prefetched (global_prefetch_b8)
//             to cover HBM latency at 1-WG/WGP occupancy.
// ---------------------------------------------------------------------------

#define KN   3072   // nodes
#define KIN  256    // in feats
#define KAH  64     // att feats / head
#define KDV  32     // v feats / head
#define KH   4      // heads

typedef unsigned short u16;
typedef __attribute__((ext_vector_type(16))) __bf16 v16bf;
typedef __attribute__((ext_vector_type(8)))  float  v8f;

union Frag16 {
  v16bf    v;
  uint32_t u[8];
  u16      h[16];
};

__device__ __forceinline__ u16 f2bfu(float f) {
  __bf16 b = (__bf16)f;                 // native v_cvt (RTE)
  u16 h;
  __builtin_memcpy(&h, &b, 2);
  return h;
}
__device__ __forceinline__ float bfu2f(u16 h) {
  return __uint_as_float(((uint32_t)h) << 16);
}
__device__ __forceinline__ void load_u16x8(uint32_t* d, const u16* s) {
  uint4 t = *(const uint4*)s;           // 16B contiguous
  d[0] = t.x; d[1] = t.y; d[2] = t.z; d[3] = t.w;
}
__device__ __forceinline__ void load_u16x16(uint32_t* d, const u16* s) {
  load_u16x8(d, s);
  load_u16x8(d + 4, s + 8);
}
__device__ __forceinline__ v8f wmma_bf16(const Frag16& a, const Frag16& b, v8f c) {
  return __builtin_amdgcn_wmma_f32_16x16x32_bf16(
      false, a.v, false, b.v, (short)0, c, false, false);
}

// ---------------------------------------------------------------------------
// Projection GEMM: out = x(3072x256) @ W(256xdout) + b, bf16 output.
// MODE 0: Q/K layout (H,N,64):  out[h][row][d],  c = h*64 + d
// MODE 1: V^T layout (H,32,N):  out[h][dv][row], c = h*32 + dv
// One 16x16 output tile per wave; 8 waves (256 threads) per block.
// ---------------------------------------------------------------------------
template <int MODE>
__global__ __launch_bounds__(256) void proj_kernel(
    const float* __restrict__ X, const float* __restrict__ W,
    const float* __restrict__ B, u16* __restrict__ out, int dout) {
  const int lane  = threadIdx.x & 31;
  const int wave  = threadIdx.x >> 5;
  const int ntile = dout >> 4;
  const int t  = blockIdx.x * 8 + wave;
  const int mt = t / ntile;
  const int nt = t % ntile;

  const int mrow  = mt * 16 + (lane & 15);          // A: M = lane%16
  const int abase = (lane >> 4) * 8;                // A half-wave K split
  const int ncol  = nt * 16 + (lane & 15);          // B: N = lane%16
  const int kbase = (lane >> 4) * 16;               // B half-wave K split

  v8f acc = {};
  for (int kk = 0; kk < KIN / 32; ++kk) {
    Frag16 a, b;
    const float* xr = X + (size_t)mrow * KIN + kk * 32;
#pragma unroll
    for (int e = 0; e < 8; ++e) a.h[e]     = f2bfu(xr[abase + e]);
#pragma unroll
    for (int e = 0; e < 8; ++e) a.h[8 + e] = f2bfu(xr[16 + abase + e]);
#pragma unroll
    for (int e = 0; e < 16; ++e)
      b.h[e] = f2bfu(W[(size_t)(kk * 32 + ((e < 8) ? (abase + e) : (8 + abase + e))) * dout + ncol]);
    acc = wmma_bf16(a, b, acc);
  }

  const float bias = B[ncol];
#pragma unroll
  for (int v = 0; v < 8; ++v) {
    const int row = mt * 16 + v + ((lane >> 4) << 3);  // C/D: M = vgpr + half*8
    const u16 val = f2bfu(acc[v] + bias);
    if (MODE == 0) {
      const int hh = ncol >> 6, d = ncol & 63;
      out[((size_t)hh * KN + row) * KAH + d] = val;
    } else {
      const int hh = ncol >> 5, dv = ncol & 31;
      out[((size_t)(hh * KDV + dv)) * KN + row] = val;
    }
  }
}

// ---------------------------------------------------------------------------
// Fused dual-branch attention. Grid: H * (N/16) blocks of 256 threads.
// Dynamic LDS: att strips (2 x 16 x 3072 bf16) + stats + output accumulator.
// ---------------------------------------------------------------------------
#define SMEM_BYTES (2 * 16 * KN * 2 + (16 + 32 + 2 * 16 * KDV) * 4)

__global__ __launch_bounds__(256) void gat_attn_kernel(
    const u16* __restrict__ q1, const u16* __restrict__ k1,
    const u16* __restrict__ q2, const u16* __restrict__ k2,
    const u16* __restrict__ vt1, const u16* __restrict__ vt2,
    const int* __restrict__ c2n, const int* __restrict__ n2c,
    const float* __restrict__ fst, const float* __restrict__ sec,
    float* __restrict__ out) {
  extern __shared__ char smem_raw[];
  u16*   att1L  = (u16*)smem_raw;               // [16][KN] bf16
  u16*   att2L  = att1L + 16 * KN;              // [16][KN] bf16
  float* statM  = (float*)(att2L + 16 * KN);    // [16] shared logit bound
  float* statS  = statM + 16;                   // [2][16] row sums
  float* outAcc = statS + 32;                   // [2][16][32] f32

  const int h    = blockIdx.x / (KN / 16);
  const int i0   = (blockIdx.x % (KN / 16)) * 16;
  const int lane = threadIdx.x & 31;
  const int wave = threadIdx.x >> 5;
  const float SCALE = 0.125f;                   // 1/sqrt(64)

  // zero row sums + output accumulator (adjacent region: 32 + 1024 floats)
  for (int t = threadIdx.x; t < 32 + 2 * 16 * KDV; t += 256) statS[t] = 0.f;

  const u16* q1h  = q1  + (size_t)h * KN * KAH;
  const u16* k1h  = k1  + (size_t)h * KN * KAH;
  const u16* q2h  = q2  + (size_t)h * KN * KAH;
  const u16* k2h  = k2  + (size_t)h * KN * KAH;
  const u16* vt1h = vt1 + (size_t)h * KDV * KN;
  const u16* vt2h = vt2 + (size_t)h * KDV * KN;

  // ---------------- phase 1: score strips att1/att2 -> LDS (bf16) ----------
  {
    const int m     = lane & 15;
    const int abase = (lane >> 4) * 8;
    const int row   = i0 + m;
    Frag16 aq1[2], aq2[2];
#pragma unroll
    for (int kk = 0; kk < 2; ++kk) {
      load_u16x16(aq1[kk].u, q1h + (size_t)row * KAH + kk * 32 + 2 * abase);
      load_u16x16(aq2[kk].u, q2h + (size_t)row * KAH + kk * 32 + 2 * abase);
    }
    const int n     = lane & 15;
    const int kbase = (lane >> 4) * 16;
    for (int jt = wave; jt < KN / 16; jt += 8) {
      const int node = jt * 16 + n;
      Frag16 b0, b1;
      // branch 1: S = Q1 (16x64) . K1^T -> 2 WMMAs over K
      load_u16x16(b0.u, k1h + (size_t)node * KAH + kbase);
      load_u16x16(b1.u, k1h + (size_t)node * KAH + 32 + kbase);
      v8f c = {};
      c = wmma_bf16(aq1[0], b0, c);
      c = wmma_bf16(aq1[1], b1, c);
#pragma unroll
      for (int v = 0; v < 8; ++v)
        att1L[(v + ((lane >> 4) << 3)) * KN + jt * 16 + n] = f2bfu(c[v] * SCALE);
      // branch 2
      load_u16x16(b0.u, k2h + (size_t)node * KAH + kbase);
      load_u16x16(b1.u, k2h + (size_t)node * KAH + 32 + kbase);
      v8f d = {};
      d = wmma_bf16(aq2[0], b0, d);
      d = wmma_bf16(aq2[1], b1, d);
#pragma unroll
      for (int v = 0; v < 8; ++v)
        att2L[(v + ((lane >> 4) << 3)) * KN + jt * 16 + n] = f2bfu(d[v] * SCALE);
    }
  }
  __syncthreads();

  // ---------------- phase 2a: LDS-only row-max bound -----------------------
  // M[i] = max_j att1[i,j] + max_j att2[i,j] >= any combined logit of either
  // branch (gather stays within row i; mask term <= 0). No global traffic.
  {
    const int row = 2 * wave + (lane >> 4);      // 2 rows per wave, 16 total
    const int jj  = lane & 15;
    const u16* a1r = att1L + row * KN;
    const u16* a2r = att2L + row * KN;
    float mx1 = -3e38f, mx2 = -3e38f;
    for (int kk = 0; kk < KN / 16; ++kk) {
      const int j = jj + 16 * kk;
      mx1 = fmaxf(mx1, bfu2f(a1r[j]));
      mx2 = fmaxf(mx2, bfu2f(a2r[j]));
    }
#pragma unroll
    for (int off = 1; off < 16; off <<= 1) {     // xor stays in 16-lane half
      mx1 = fmaxf(mx1, __shfl_xor(mx1, off, 32));
      mx2 = fmaxf(mx2, __shfl_xor(mx2, off, 32));
    }
    if (jj == 0) statM[row] = mx1 + mx2;
  }
  __syncthreads();

  // ---------------- phase 2b: P @ V via WMMA + fused row sums --------------
  {
    const int b  = wave & 1;                     // branch per wave
    const int wp = wave >> 1;                    // 4 waves per branch
    const u16* attA = b ? att2L : att1L;
    const u16* attB = b ? att1L : att2L;
    const int* idx  = b ? n2c : c2n;
    const float* gr = b ? sec : fst;
    const u16* vth  = b ? vt2h : vt1h;

    const int m     = lane & 15;
    const int abase = (lane >> 4) * 8;
    const int n     = lane & 15;
    const int kbase = (lane >> 4) * 16;
    const float mrow = statM[m];
    const int* idxR = idx + (size_t)(i0 + m) * KN;
    const u16* aAr  = attA + m * KN;
    const u16* aBr  = attB + m * KN;

    v8f acc0 = {}, acc1 = {};
    float psum = 0.f;                            // this lane's disjoint j's
    for (int ch = wp; ch < KN / 32; ch += 4) {
      // prefetch next chunk's tiles: 32 graph lines (lane -> one j row) and
      // 16 index row-chunks (lane -> half of one row). global_prefetch_b8.
      if (ch + 4 < KN / 32) {
        const int jn = (ch + 4) * 32;
        __builtin_prefetch(gr + (size_t)(jn + lane) * KN + i0, 0, 3);
        __builtin_prefetch(idx + (size_t)(i0 + (lane & 15)) * KN + jn +
                               ((lane >> 4) << 4), 0, 3);
      }
      Frag16 p;
#pragma unroll
      for (int e = 0; e < 16; ++e) {
        const int j = ch * 32 + 2 * abase + e;
        const float l = bfu2f(aAr[j]) + bfu2f(aBr[idxR[j]]) +
                        (gr[(size_t)j * KN + (i0 + m)] - 1.f) * 10000.f;
        const float pe = __expf(l - mrow);       // <= 1 by construction
        psum += pe;
        p.h[e] = f2bfu(pe);
      }
      Frag16 b0, b1;   // V^T rows = output dims; contiguous along j
      load_u16x16(b0.u, vth + (size_t)n * KN + ch * 32 + kbase);
      load_u16x16(b1.u, vth + (size_t)(16 + n) * KN + ch * 32 + kbase);
      acc0 = wmma_bf16(p, b0, acc0);
      acc1 = wmma_bf16(p, b1, acc1);
    }
    atomicAdd(&statS[b * 16 + m], psum);         // exact row denominator
    float* oa = outAcc + b * (16 * KDV);
#pragma unroll
    for (int v = 0; v < 8; ++v) {
      const int mr = v + ((lane >> 4) << 3);
      atomicAdd(&oa[mr * KDV + n], acc0[v]);
      atomicAdd(&oa[mr * KDV + 16 + n], acc1[v]);
    }
  }
  __syncthreads();

  // ---------------- finalize: divide by softmax sum, write (N,256) ---------
  for (int t = threadIdx.x; t < 2 * 16 * KDV; t += 256) {
    const int b = t >> 9, rem = t & 511, m = rem >> 5, dv = rem & 31;
    const float val = outAcc[t] / statS[b * 16 + m];
    out[(size_t)(i0 + m) * 256 + b * 128 + h * KDV + dv] = val;
  }
}

// ---------------------------------------------------------------------------
extern "C" void kernel_launch(void* const* d_in, const int* in_sizes, int n_in,
                              void* d_out, int out_size, void* d_ws, size_t ws_size,
                              hipStream_t stream) {
  const float* x   = (const float*)d_in[0];
  const float* fst = (const float*)d_in[1];
  const float* sec = (const float*)d_in[2];
  const int*   n2c = (const int*)d_in[3];
  const int*   c2n = (const int*)d_in[4];
  const float* Wq1 = (const float*)d_in[5];  const float* bq1 = (const float*)d_in[6];
  const float* Wk1 = (const float*)d_in[7];  const float* bk1 = (const float*)d_in[8];
  const float* Wv1 = (const float*)d_in[9];  const float* bv1 = (const float*)d_in[10];
  const float* Wq2 = (const float*)d_in[11]; const float* bq2 = (const float*)d_in[12];
  const float* Wk2 = (const float*)d_in[13]; const float* bk2 = (const float*)d_in[14];
  const float* Wv2 = (const float*)d_in[15]; const float* bv2 = (const float*)d_in[16];
  float* out = (float*)d_out;

  // workspace carve (bf16): 4x Q/K (H*N*64) + 2x V^T (H*32*N) = 7.5 MB
  const size_t qkElems = (size_t)KH * KN * KAH;   // 786432
  const size_t vtElems = (size_t)KH * KDV * KN;   // 393216
  u16* q1  = (u16*)d_ws;
  u16* k1  = q1  + qkElems;
  u16* q2  = k1  + qkElems;
  u16* k2  = q2  + qkElems;
  u16* vt1 = k2  + qkElems;
  u16* vt2 = vt1 + vtElems;

  // projections: 192*(dout/16) tiles, 8 tiles (waves) per block
  proj_kernel<0><<<(KN / 16) * (256 / 16) / 8, 256, 0, stream>>>(x, Wq1, bq1, q1, 256);
  proj_kernel<0><<<(KN / 16) * (256 / 16) / 8, 256, 0, stream>>>(x, Wk1, bk1, k1, 256);
  proj_kernel<0><<<(KN / 16) * (256 / 16) / 8, 256, 0, stream>>>(x, Wq2, bq2, q2, 256);
  proj_kernel<0><<<(KN / 16) * (256 / 16) / 8, 256, 0, stream>>>(x, Wk2, bk2, k2, 256);
  proj_kernel<1><<<(KN / 16) * (128 / 16) / 8, 256, 0, stream>>>(x, Wv1, bv1, vt1, 128);
  proj_kernel<1><<<(KN / 16) * (128 / 16) / 8, 256, 0, stream>>>(x, Wv2, bv2, vt2, 128);

  // fused attention: one block per (head, 16-row strip)
  gat_attn_kernel<<<KH * (KN / 16), 256, SMEM_BYTES, stream>>>(
      q1, k1, q2, k2, vt1, vt2, c2n, n2c, fst, sec, out);
}
